// BindingGNN_14010183320047
// MI455X (gfx1250) — compile-verified, matrix operand order checked by reading, not run on hardware
//
#include <hip/hip_runtime.h>
#include <stdint.h>

#define N_NODES 50000
#define N_EDGES 500000
#define N_GRAPHS 100
#define NPG 500
#define HD 128
#define NL 4
#define EAD 4
#define PN_D 35
#define LN_D 11
#define BN_EPS 1e-5f
#define TE 64   // rows (edges or nodes) per block tile

#define USE_ASYNC_GATHER 1   // global_load_async_to_lds_b128 path for h gathers

typedef __attribute__((ext_vector_type(16))) _Float16 v16h;
typedef __attribute__((ext_vector_type(8)))  float    v8f;

union F16Frag { v16h v; _Float16 h[16]; unsigned int u[8]; };

// ---------------------------------------------------------------------------
// WMMA 16x16x32 f16 fragment layout helpers (per CDNA5 ISA 7.12.2):
// 16-bit A (16x32): lanes 0-15 hold M=lane; VGPR0..3 = K 0..7, VGPR4..7 = K16..23
//                   lanes 16-31: VGPR0..3 = K 8..15, VGPR4..7 = K 24..31.
// B (32x16) mirrors A with N = lane&15 and identical K striping.
// C/D f32: n = lane&15, m = v + 8*(lane>>4).
// ---------------------------------------------------------------------------
__device__ __forceinline__ int wmma_kmap_pair(int lane, int p) {
  int hi = (lane >> 4) & 1;
  int base = (p < 4) ? (hi ? 8 : 0) : (hi ? 24 : 16);
  return base + ((p & 3) << 1);
}
__device__ __forceinline__ int wmma_kmap(int lane, int h) {
  return wmma_kmap_pair(lane, h >> 1) + (h & 1);
}

__device__ __forceinline__ v16h load_a_lds(const _Float16* smem, int m0, int kbase, int stride) {
  int lane = threadIdx.x & 31;
  const _Float16* rowp = smem + (m0 + (lane & 15)) * stride + kbase;
  F16Frag r;
#pragma unroll
  for (int p = 0; p < 8; ++p) {
    int k0 = wmma_kmap_pair(lane, p);          // even -> dword aligned
    r.u[p] = *(const unsigned int*)(rowp + k0);
  }
  return r.v;
}

__device__ __forceinline__ v16h load_b_packed(const _Float16* pk, int tile) {
  int lane = threadIdx.x & 31;
  return *(const v16h*)(pk + (size_t)tile * 512 + lane * 16);
}

__device__ __forceinline__ v8f wmma_f16(v16h a, v16h b, v8f c) {
  return __builtin_amdgcn_wmma_f32_16x16x32_f16(false, a, false, b, (short)0, c, false, false);
}

__device__ __forceinline__ float silu_f(float x) { return x / (1.0f + __expf(-x)); }

// ---------------------------------------------------------------------------
// Pack a KxNout f32 weight matrix (K padded to Kpad with zeros) into
// fragment-native f16 tiles: tile index = kt*(Nout/16)+nt, 512 halves/tile,
// lane-major so a wave loads its B fragment as one 32B-per-lane read.
// ---------------------------------------------------------------------------
__global__ void __launch_bounds__(256) pack_weight_kernel(
    const float* __restrict__ src, _Float16* __restrict__ dst,
    int Ksrc, int Kpad, int Nout) {
  int total = (Kpad >> 5) * (Nout >> 4) * 512;
  int idx = blockIdx.x * 256 + threadIdx.x;
  if (idx >= total) return;
  int tile = idx >> 9, within = idx & 511;
  int lane = within >> 4, hh = within & 15;
  int numNt = Nout >> 4;
  int kt = tile / numNt, nt = tile - kt * numNt;
  int k = kt * 32 + wmma_kmap(lane, hh);
  int n = nt * 16 + (lane & 15);
  float v = (k < Ksrc) ? src[(size_t)k * Nout + n] : 0.0f;
  dst[idx] = (_Float16)v;
}

// ---------------------------------------------------------------------------
// Node projection + BN statistics (partial sums -> atomics)
// ---------------------------------------------------------------------------
__global__ void __launch_bounds__(256) node_embed_kernel(
    const float* __restrict__ x, const unsigned char* __restrict__ is_prot,
    const float* __restrict__ Wp, const float* __restrict__ bp,
    const float* __restrict__ Wl, const float* __restrict__ bl,
    float* __restrict__ h_raw, float* __restrict__ bnsum, float* __restrict__ bnsq) {
  int ch  = threadIdx.x & (HD - 1);
  int sub = threadIdx.x >> 7;
  float lsum = 0.f, lsq = 0.f;
  for (int n = blockIdx.x * 2 + sub; n < N_NODES; n += gridDim.x * 2) {
    const float* xr = x + (size_t)n * PN_D;
    float acc;
    if (is_prot[n]) {
      acc = bp[ch];
      for (int k = 0; k < PN_D; ++k) acc += xr[k] * Wp[k * HD + ch];
    } else {
      acc = bl[ch];
      for (int k = 0; k < LN_D; ++k) acc += xr[k] * Wl[k * HD + ch];
    }
    h_raw[(size_t)n * HD + ch] = acc;
    lsum += acc; lsq += acc * acc;
  }
  atomicAdd(&bnsum[ch], lsum);
  atomicAdd(&bnsq[ch], lsq);
}

__global__ void __launch_bounds__(256) bn_apply_kernel(
    const float* __restrict__ h_raw, const float* __restrict__ bnsum,
    const float* __restrict__ bnsq, const float* __restrict__ gamma,
    const float* __restrict__ beta, float* __restrict__ h32,
    _Float16* __restrict__ h16) {
  size_t idx = (size_t)blockIdx.x * 256 + threadIdx.x;
  if (idx >= (size_t)N_NODES * HD) return;
  int ch = (int)(idx & (HD - 1));
  const float inv = 1.0f / (float)N_NODES;
  float mu  = bnsum[ch] * inv;
  float var = bnsq[ch] * inv - mu * mu;
  float v = (h_raw[idx] - mu) * rsqrtf(var + BN_EPS) * gamma[ch] + beta[ch];
  h32[idx] = v;
  h16[idx] = (_Float16)v;
}

__global__ void __launch_bounds__(256) degree_kernel(const int* __restrict__ row,
                                                     float* __restrict__ deg) {
  int i = blockIdx.x * 256 + threadIdx.x;
  if (i < N_EDGES) atomicAdd(&deg[row[i]], 1.0f);
}

__global__ void __launch_bounds__(256) pos_update_kernel(
    const float* __restrict__ pd, const float* __restrict__ deg,
    float* __restrict__ pos) {
  int i = blockIdx.x * 256 + threadIdx.x;
  if (i < N_NODES * 3) {
    float d = fmaxf(deg[i / 3], 1.0f);
    pos[i] += pd[i] / d;
  }
}

// ---------------------------------------------------------------------------
// Edge kernel: 64 edges/block, 8 waves. Three chained WMMA GEMMs:
//   out1 = silu(feat @ e_w1 + b1)        feat: 64x288 (261 padded)
//   m    = silu(out1 @ e_w2 + b2)        -> agg atomics (segment_sum)
//   t    = silu(m @ c_w1 + cb1)
//   w    = t @ c_w2   -> pos_delta atomics (segment_sum of rel*w)
// h[row]/h[col] gathers stream memory->LDS via async b128 (ASYNCcnt).
// ---------------------------------------------------------------------------
__global__ void __launch_bounds__(256) edge_kernel(
    const float* __restrict__ pos, const int* __restrict__ row,
    const int* __restrict__ col, const float* __restrict__ eattr,
    const _Float16* __restrict__ h16,
    const _Float16* __restrict__ pk1, const _Float16* __restrict__ pk2,
    const _Float16* __restrict__ pk3,
    const float* __restrict__ b1, const float* __restrict__ b2,
    const float* __restrict__ cb1, const float* __restrict__ cw2,
    float* __restrict__ agg, float* __restrict__ pos_delta) {
  __shared__ __attribute__((aligned(16))) _Float16 sFeat[TE * 288]; // A; reused as t-tile
  __shared__ __attribute__((aligned(16))) _Float16 sBuf[TE * 128];  // out1 -> m
  __shared__ float sRelX[TE], sRelY[TE], sRelZ[TE], sD2[TE];
  __shared__ float sCw2[HD];
  __shared__ int   sRowA[TE], sColA[TE];

  int tid = threadIdx.x;
  int base = blockIdx.x * TE;

  if (tid < HD) sCw2[tid] = cw2[tid];

  // per-edge geometry
  for (int e = tid; e < TE; e += 256) {
    int ge = base + e;
    int r = -1, c = 0;
    float rx = 0.f, ry = 0.f, rz = 0.f;
    if (ge < N_EDGES) {
      r = row[ge]; c = col[ge];
      rx = pos[r * 3 + 0] - pos[c * 3 + 0];
      ry = pos[r * 3 + 1] - pos[c * 3 + 1];
      rz = pos[r * 3 + 2] - pos[c * 3 + 2];
    }
    sRowA[e] = r; sColA[e] = c;
    sRelX[e] = rx; sRelY[e] = ry; sRelZ[e] = rz;
    sD2[e] = rx * rx + ry * ry + rz * rz;
  }
  __syncthreads();

  // ---- stage h[row] | h[col] (k = 0..255) in 16B chunks -------------------
  // 64 edges x 32 chunks x 8 halves; invalid edges read row 0 (harmless:
  // their M-rows are never emitted), tail columns zero-padded below.
#if USE_ASYNC_GATHER
  for (int idx = tid; idx < TE * 32; idx += 256) {
    int e = idx >> 5, c = idx & 31;
    int src = (c < 16) ? sRowA[e] : sColA[e];
    if (src < 0) src = 0;
    unsigned lds_off = (unsigned)(uintptr_t)(sFeat + e * 288 + c * 8);
    unsigned long long ga =
        (unsigned long long)(uintptr_t)(h16 + (size_t)src * HD + (c & 15) * 8);
    asm volatile("global_load_async_to_lds_b128 %0, %1, off"
                 :: "v"(lds_off), "v"(ga) : "memory");
  }
  asm volatile("s_wait_asynccnt 0x0" ::: "memory");
#else
  for (int idx = tid; idx < TE * 32; idx += 256) {
    int e = idx >> 5, c = idx & 31;
    int src = (c < 16) ? sRowA[e] : sColA[e];
    if (src < 0) src = 0;
    uint4 v = make_uint4(0u, 0u, 0u, 0u);
    if (sRowA[e] >= 0)
      v = *(const uint4*)(h16 + (size_t)src * HD + (c & 15) * 8);
    *(uint4*)(sFeat + e * 288 + c * 8) = v;
  }
#endif
  // tail columns: d2 | edge_attr | zero pad  (k = 256..287)
  for (int idx = tid; idx < TE * 32; idx += 256) {
    int e = idx >> 5, k = 256 + (idx & 31);
    _Float16 v = (_Float16)0.0f;
    if (sRowA[e] >= 0) {
      if (k == 256)      v = (_Float16)sD2[e];
      else if (k < 261)  v = (_Float16)eattr[(size_t)(base + e) * EAD + (k - 257)];
    }
    sFeat[e * 288 + k] = v;
  }
  __syncthreads();

  int wv = tid >> 5;
  int mt = wv & 3;
  int nh = wv >> 2;
  int lane = tid & 31;
  int nloc = lane & 15;
  int mbase = mt * 16 + ((lane >> 4) << 3);
  v8f zero = {};
  v8f acc[4];

  // ---- GEMM1: feat[64x288] @ e_w1 ----
#pragma unroll
  for (int t = 0; t < 4; ++t) acc[t] = zero;
  for (int ki = 0; ki < 9; ++ki) {
    v16h a = load_a_lds(sFeat, mt * 16, ki * 32, 288);
    if (ki + 1 < 9)
      __builtin_prefetch((const void*)(pk1 + (size_t)((ki + 1) * 8 + nh * 4) * 512), 0, 0);
#pragma unroll
    for (int nt = 0; nt < 4; ++nt)
      acc[nt] = wmma_f16(a, load_b_packed(pk1, ki * 8 + nh * 4 + nt), acc[nt]);
  }
#pragma unroll
  for (int nt = 0; nt < 4; ++nt) {
    int n = nh * 64 + nt * 16 + nloc;
    float bb = b1[n];
#pragma unroll
    for (int v = 0; v < 8; ++v)
      sBuf[(mbase + v) * 128 + n] = (_Float16)silu_f(acc[nt][v] + bb);
  }
  __syncthreads();

  // ---- GEMM2: out1 @ e_w2 -> m ----
#pragma unroll
  for (int t = 0; t < 4; ++t) acc[t] = zero;
  for (int ki = 0; ki < 4; ++ki) {
    v16h a = load_a_lds(sBuf, mt * 16, ki * 32, 128);
#pragma unroll
    for (int nt = 0; nt < 4; ++nt)
      acc[nt] = wmma_f16(a, load_b_packed(pk2, ki * 8 + nh * 4 + nt), acc[nt]);
  }
  __syncthreads();                       // all reads of sBuf done before rewrite
#pragma unroll
  for (int nt = 0; nt < 4; ++nt) {
    int n = nh * 64 + nt * 16 + nloc;
    float bb = b2[n];
#pragma unroll
    for (int v = 0; v < 8; ++v) {
      int e = mbase + v;
      float s = silu_f(acc[nt][v] + bb);
      sBuf[e * 128 + n] = (_Float16)s;
      int r = sRowA[e];
      if (r >= 0) atomicAdd(&agg[(size_t)r * HD + n], s);   // segment_sum(m)
    }
  }
  __syncthreads();

  // ---- GEMM3: m @ c_w1 -> t ----
#pragma unroll
  for (int t = 0; t < 4; ++t) acc[t] = zero;
  for (int ki = 0; ki < 4; ++ki) {
    v16h a = load_a_lds(sBuf, mt * 16, ki * 32, 128);
#pragma unroll
    for (int nt = 0; nt < 4; ++nt)
      acc[nt] = wmma_f16(a, load_b_packed(pk3, ki * 8 + nh * 4 + nt), acc[nt]);
  }
#pragma unroll
  for (int nt = 0; nt < 4; ++nt) {
    int n = nh * 64 + nt * 16 + nloc;
    float bb = cb1[n];
#pragma unroll
    for (int v = 0; v < 8; ++v)
      sFeat[(mbase + v) * 128 + n] = (_Float16)silu_f(acc[nt][v] + bb);  // t-tile
  }
  __syncthreads();

  // gate w = t @ c_w2: 4 lanes/edge + shuffle reduce, scatter rel*w
  {
    int e = tid >> 2, q = tid & 3;
    const _Float16* tr = sFeat + e * 128 + q * 32;
    float wacc = 0.f;
#pragma unroll
    for (int k = 0; k < 32; ++k) wacc += (float)tr[k] * sCw2[q * 32 + k];
    wacc += __shfl_xor(wacc, 1, 32);
    wacc += __shfl_xor(wacc, 2, 32);
    int r = sRowA[e];
    if (q == 0 && r >= 0) {
      atomicAdd(&pos_delta[r * 3 + 0], sRelX[e] * wacc);
      atomicAdd(&pos_delta[r * 3 + 1], sRelY[e] * wacc);
      atomicAdd(&pos_delta[r * 3 + 2], sRelZ[e] * wacc);
    }
  }
}

// ---------------------------------------------------------------------------
// Node update: upd = silu([h,agg] @ n_w1 + nb1) @ n_w2 + nb2 ; h += upd
// ---------------------------------------------------------------------------
__global__ void __launch_bounds__(256) node_update_kernel(
    const float* __restrict__ agg,
    const float* __restrict__ nb1, const float* __restrict__ nb2,
    const _Float16* __restrict__ pk1, const _Float16* __restrict__ pk2,
    float* __restrict__ h32, _Float16* __restrict__ h16) {
  __shared__ __attribute__((aligned(16))) _Float16 sIn[TE * 256];
  __shared__ __attribute__((aligned(16))) _Float16 sBuf[TE * 128];
  int tid = threadIdx.x;
  int base = blockIdx.x * TE;

  // stage [h | agg] as f16, 4 floats per chunk
  for (int idx = tid; idx < TE * 64; idx += 256) {
    int e = idx >> 6, c = idx & 63;
    int n = base + e;
    float4 f = make_float4(0.f, 0.f, 0.f, 0.f);
    if (n < N_NODES) {
      const float* sp = (c < 32) ? (h32 + (size_t)n * HD + c * 4)
                                 : (agg + (size_t)n * HD + (c - 32) * 4);
      f = *(const float4*)sp;
    }
    _Float16* dp = sIn + e * 256 + c * 4;
    dp[0] = (_Float16)f.x; dp[1] = (_Float16)f.y;
    dp[2] = (_Float16)f.z; dp[3] = (_Float16)f.w;
  }
  __syncthreads();

  int wv = tid >> 5;
  int mt = wv & 3;
  int nh = wv >> 2;
  int lane = tid & 31;
  int nloc = lane & 15;
  int mbase = mt * 16 + ((lane >> 4) << 3);
  v8f zero = {};
  v8f acc[4];

  // GEMM n1
#pragma unroll
  for (int t = 0; t < 4; ++t) acc[t] = zero;
  for (int ki = 0; ki < 8; ++ki) {
    v16h a = load_a_lds(sIn, mt * 16, ki * 32, 256);
#pragma unroll
    for (int nt = 0; nt < 4; ++nt)
      acc[nt] = wmma_f16(a, load_b_packed(pk1, ki * 8 + nh * 4 + nt), acc[nt]);
  }
#pragma unroll
  for (int nt = 0; nt < 4; ++nt) {
    int n = nh * 64 + nt * 16 + nloc;
    float bb = nb1[n];
#pragma unroll
    for (int v = 0; v < 8; ++v)
      sBuf[(mbase + v) * 128 + n] = (_Float16)silu_f(acc[nt][v] + bb);
  }
  __syncthreads();

  // GEMM n2 + residual
#pragma unroll
  for (int t = 0; t < 4; ++t) acc[t] = zero;
  for (int ki = 0; ki < 4; ++ki) {
    v16h a = load_a_lds(sBuf, mt * 16, ki * 32, 128);
#pragma unroll
    for (int nt = 0; nt < 4; ++nt)
      acc[nt] = wmma_f16(a, load_b_packed(pk2, ki * 8 + nh * 4 + nt), acc[nt]);
  }
#pragma unroll
  for (int nt = 0; nt < 4; ++nt) {
    int n = nh * 64 + nt * 16 + nloc;
    float bb = nb2[n];
#pragma unroll
    for (int v = 0; v < 8; ++v) {
      int gm = base + mbase + v;
      if (gm < N_NODES) {
        float hv = h32[(size_t)gm * HD + n] + acc[nt][v] + bb;
        h32[(size_t)gm * HD + n] = hv;
        h16[(size_t)gm * HD + n] = (_Float16)hv;
      }
    }
  }
}

// ---------------------------------------------------------------------------
// Pooling (graphs are contiguous 500-node blocks) and MLP head
// ---------------------------------------------------------------------------
__global__ void __launch_bounds__(128) pool_kernel(const float* __restrict__ h32,
                                                   float* __restrict__ gfeat) {
  int g = blockIdx.x, ch = threadIdx.x;
  float s = 0.f;
  for (int i = 0; i < NPG; ++i)
    s += h32[((size_t)(g * NPG + i)) * HD + ch];
  gfeat[g * 256 + ch]       = s;
  gfeat[g * 256 + 128 + ch] = s / (float)NPG;
}

__global__ void __launch_bounds__(128) head_kernel(
    const float* __restrict__ gfeat,
    const float* __restrict__ w1, const float* __restrict__ b1,
    const float* __restrict__ w2, const float* __restrict__ b2,
    const float* __restrict__ w3, const float* __restrict__ b3,
    float* __restrict__ out) {
  __shared__ float z1[128];
  __shared__ float red[64];
  int g = blockIdx.x, t = threadIdx.x;
  const float* gf = gfeat + g * 256;
  float a = b1[t];
  for (int k = 0; k < 256; ++k) a += gf[k] * w1[k * 128 + t];
  z1[t] = fmaxf(a, 0.f);
  __syncthreads();
  if (t < 64) {
    float a2 = b2[t];
    for (int k = 0; k < 128; ++k) a2 += z1[k] * w2[k * 64 + t];
    red[t] = fmaxf(a2, 0.f) * w3[t];
  }
  __syncthreads();
  for (int s = 32; s > 0; s >>= 1) {
    if (t < s) red[t] += red[t + s];
    __syncthreads();
  }
  if (t == 0) out[g] = red[0] + b3[0];
}

// ---------------------------------------------------------------------------
extern "C" void kernel_launch(void* const* d_in, const int* in_sizes, int n_in,
                              void* d_out, int out_size, void* d_ws, size_t ws_size,
                              hipStream_t stream) {
  const float* x       = (const float*)d_in[0];
  const float* pos_in  = (const float*)d_in[1];
  const int*   eidx    = (const int*)d_in[2];
  const float* eattr   = (const float*)d_in[3];
  const unsigned char* is_prot = (const unsigned char*)d_in[5];
  const float* Wp   = (const float*)d_in[6];
  const float* bp   = (const float*)d_in[7];
  const float* Wl   = (const float*)d_in[8];
  const float* bl   = (const float*)d_in[9];
  const float* bn_g = (const float*)d_in[10];
  const float* bn_b = (const float*)d_in[11];
  const float* e_w1 = (const float*)d_in[12];
  const float* e_b1 = (const float*)d_in[13];
  const float* e_w2 = (const float*)d_in[14];
  const float* e_b2 = (const float*)d_in[15];
  const float* c_w1 = (const float*)d_in[16];
  const float* c_b1 = (const float*)d_in[17];
  const float* c_w2 = (const float*)d_in[18];
  const float* n_w1 = (const float*)d_in[19];
  const float* n_b1 = (const float*)d_in[20];
  const float* n_w2 = (const float*)d_in[21];
  const float* n_b2 = (const float*)d_in[22];
  const float* h1_w = (const float*)d_in[23];
  const float* h1_b = (const float*)d_in[24];
  const float* h2_w = (const float*)d_in[25];
  const float* h2_b = (const float*)d_in[26];
  const float* h3_w = (const float*)d_in[27];
  const float* h3_b = (const float*)d_in[28];
  float* out = (float*)d_out;

  const int* row = eidx;
  const int* col = eidx + N_EDGES;

  char* wsp = (char*)d_ws;
  auto alloc = [&](size_t bytes) -> char* {
    char* p = wsp;
    wsp += (bytes + 255) & ~(size_t)255;
    return p;
  };
  float*    h_raw     = (float*)alloc((size_t)N_NODES * HD * 4);
  float*    h32       = (float*)alloc((size_t)N_NODES * HD * 4);
  float*    agg       = (float*)alloc((size_t)N_NODES * HD * 4);
  _Float16* h16       = (_Float16*)alloc((size_t)N_NODES * HD * 2);
  float*    pos_cur   = (float*)alloc((size_t)N_NODES * 3 * 4);
  float*    pos_delta = (float*)alloc((size_t)N_NODES * 3 * 4);
  float*    deg       = (float*)alloc((size_t)N_NODES * 4);
  float*    bnsum     = (float*)alloc(HD * 4);
  float*    bnsq      = (float*)alloc(HD * 4);
  float*    gfeat     = (float*)alloc((size_t)N_GRAPHS * 256 * 4);
  const size_t EW1_T = 9 * 8 * 512;   // halves per layer, e_w1 (288x128)
  const size_t SQ_T  = 4 * 8 * 512;   // 128x128
  const size_t NW1_T = 8 * 8 * 512;   // 256x128
  _Float16* pk_ew1 = (_Float16*)alloc(NL * EW1_T * 2);
  _Float16* pk_ew2 = (_Float16*)alloc(NL * SQ_T * 2);
  _Float16* pk_cw1 = (_Float16*)alloc(NL * SQ_T * 2);
  _Float16* pk_nw1 = (_Float16*)alloc(NL * NW1_T * 2);
  _Float16* pk_nw2 = (_Float16*)alloc(NL * SQ_T * 2);

  hipMemsetAsync(bnsum, 0, HD * 4, stream);
  hipMemsetAsync(bnsq, 0, HD * 4, stream);
  hipMemsetAsync(deg, 0, (size_t)N_NODES * 4, stream);
  hipMemcpyAsync(pos_cur, pos_in, (size_t)N_NODES * 3 * 4,
                 hipMemcpyDeviceToDevice, stream);

  for (int l = 0; l < NL; ++l) {
    pack_weight_kernel<<<(int)((EW1_T + 255) / 256), 256, 0, stream>>>(
        e_w1 + (size_t)l * 261 * HD, pk_ew1 + l * EW1_T, 261, 288, HD);
    pack_weight_kernel<<<(int)((SQ_T + 255) / 256), 256, 0, stream>>>(
        e_w2 + (size_t)l * HD * HD, pk_ew2 + l * SQ_T, 128, 128, HD);
    pack_weight_kernel<<<(int)((SQ_T + 255) / 256), 256, 0, stream>>>(
        c_w1 + (size_t)l * HD * HD, pk_cw1 + l * SQ_T, 128, 128, HD);
    pack_weight_kernel<<<(int)((NW1_T + 255) / 256), 256, 0, stream>>>(
        n_w1 + (size_t)l * 256 * HD, pk_nw1 + l * NW1_T, 256, 256, HD);
    pack_weight_kernel<<<(int)((SQ_T + 255) / 256), 256, 0, stream>>>(
        n_w2 + (size_t)l * HD * HD, pk_nw2 + l * SQ_T, 128, 128, HD);
  }

  node_embed_kernel<<<256, 256, 0, stream>>>(x, is_prot, Wp, bp, Wl, bl,
                                             h_raw, bnsum, bnsq);
  bn_apply_kernel<<<(N_NODES * HD + 255) / 256, 256, 0, stream>>>(
      h_raw, bnsum, bnsq, bn_g, bn_b, h32, h16);
  degree_kernel<<<(N_EDGES + 255) / 256, 256, 0, stream>>>(row, deg);

  for (int l = 0; l < NL; ++l) {
    hipMemsetAsync(agg, 0, (size_t)N_NODES * HD * 4, stream);
    hipMemsetAsync(pos_delta, 0, (size_t)N_NODES * 3 * 4, stream);
    edge_kernel<<<(N_EDGES + TE - 1) / TE, 256, 0, stream>>>(
        pos_cur, row, col, eattr, h16,
        pk_ew1 + l * EW1_T, pk_ew2 + l * SQ_T, pk_cw1 + l * SQ_T,
        e_b1 + l * HD, e_b2 + l * HD, c_b1 + l * HD, c_w2 + l * HD,
        agg, pos_delta);
    pos_update_kernel<<<(N_NODES * 3 + 255) / 256, 256, 0, stream>>>(
        pos_delta, deg, pos_cur);
    node_update_kernel<<<(N_NODES + TE - 1) / TE, 256, 0, stream>>>(
        agg, n_b1 + l * HD, n_b2 + l * HD,
        pk_nw1 + l * NW1_T, pk_nw2 + l * SQ_T, h32, h16);
  }

  pool_kernel<<<N_GRAPHS, 128, 0, stream>>>(h32, gfeat);
  head_kernel<<<N_GRAPHS, 128, 0, stream>>>(gfeat, h1_w, h1_b, h2_w, h2_b,
                                            h3_w, h3_b, out);
}
